// AlphaComplexClassifier_12506944766309
// MI455X (gfx1250) — matrix operand order repaired and checked by reading, not compiled, not run on hardware
//
#include <hip/hip_runtime.h>
#include <cmath>

typedef __attribute__((ext_vector_type(16))) _Float16 v16h;
typedef __attribute__((ext_vector_type(8)))  _Float16 v8h;
typedef __attribute__((ext_vector_type(8)))  float    v8f;

#define TN 4096   // tokens
#define NV 2048   // vertices

// ---------------- constexpr GA tables ----------------
constexpr int   GRADE_[16] = {0,1,1,1,1,2,2,2,2,2,2,3,3,3,3,4};
constexpr float INVM_[16]  = {1.f,0.f,1.f,1.f,1.f,0.f,0.f,0.f,1.f,1.f,1.f,0.f,0.f,0.f,1.f,0.f};
// blade index -> bitmask (bit0=e0..bit3=e3), and inverse
constexpr int BMASK_[16] = {0,1,2,4,8,3,5,9,6,10,12,7,11,13,14,15};
constexpr int M2I_[16]   = {0,1,2,5,3,6,8,11,4,7,9,12,10,13,14,15};

constexpr int popc4_(int x){ int c=0; for(int b=0;b<4;++b) c += (x>>b)&1; return c; }
constexpr int rsign_(int a, int b){ int s=0; int t=a>>1; while(t){ s += popc4_(t & b); t >>= 1; } return (s&1)?-1:1; }

// packed-weight offsets (halfs) within one block's pack region (total 50688)
#define PK_WQ 0
#define PK_WK 2560
#define PK_WV 5120
#define PK_WO 7680
#define PK_W1 10240
#define PK_W2 15360
#define PK_SQ 17920
#define PK_SK 22016
#define PK_SV 26112
#define PK_SO 30208
#define PK_S1 34304
#define PK_S2 42496
#define PK_PER_BLOCK 50688

struct BlockP {
  const float *wq,*wk,*wv,*wo,*sq,*sk,*sv,*so,*so_b,*w1,*w2,*s1,*s1_b,*s2,*s2_b,*ln1_g,*ln1_b,*ln2_g,*ln2_b;
};

// ---------------- WMMA helpers ----------------
__device__ __forceinline__ v8f wmma32(v16h a, v16h b, v8f c){
  return __builtin_amdgcn_wmma_f32_16x16x32_f16(false, a, false, b, (short)0, c, false, false);
}
__device__ __forceinline__ v16h combine8(v8h lo, v8h hi){
  v16h r;
#pragma unroll
  for(int i=0;i<8;++i){ r[i]=lo[i]; r[i+8]=hi[i]; }
  return r;
}
// A-fragment (16x32 f16): lane l holds row M=l&15; K = koff + i + 16*(l>>4)
__device__ __forceinline__ v16h loadA16(const _Float16* base, int ld, int koff){
  const int l = (int)(threadIdx.x & 31u);
  const _Float16* r = base + (size_t)(l & 15) * (size_t)ld + (size_t)(koff + ((l >> 4) << 4));
  return combine8(*(const v8h*)r, *(const v8h*)(r + 8));
}
// B-fragment (32x16 f16): lane l holds col N=l&15; K = koff + i + 16*(l>>4)
__device__ __forceinline__ v16h loadB16(const _Float16* base, size_t ld, int koff){
  const int l = (int)(threadIdx.x & 31u);
  const _Float16* r = base + (size_t)(l & 15) * ld + (size_t)(koff + ((l >> 4) << 4));
  return combine8(*(const v8h*)r, *(const v8h*)(r + 8));
}
// B-fragment from pre-swizzled pack: lane l reads halfs [l*16 .. l*16+15]
__device__ __forceinline__ v16h loadBpk(const _Float16* p){
  const int l = (int)(threadIdx.x & 31u);
  const _Float16* r = p + l*16;
  return combine8(*(const v8h*)r, *(const v8h*)(r + 8));
}
__device__ __forceinline__ float redmax16(float v){
  v = fmaxf(v, __shfl_xor(v, 1)); v = fmaxf(v, __shfl_xor(v, 2));
  v = fmaxf(v, __shfl_xor(v, 4)); v = fmaxf(v, __shfl_xor(v, 8));
  return v;
}
__device__ __forceinline__ float redsum16(float v){
  v += __shfl_xor(v, 1); v += __shfl_xor(v, 2);
  v += __shfl_xor(v, 4); v += __shfl_xor(v, 8);
  return v;
}
__device__ __forceinline__ float gelu_(float v){
  float v3 = v*v*v;
  return 0.5f*v*(1.f + tanhf(0.7978845608f*(v + 0.044715f*v3)));
}

// geometric product of Cl(3,0,1): fully unrolled from constexpr Cayley table
__device__ __forceinline__ void geo_product(const float* A, const float* B, float* U){
#pragma unroll
  for(int i=0;i<16;++i){
#pragma unroll
    for(int j=0;j<16;++j){
      const int ma = BMASK_[i], mb = BMASK_[j];
      if((ma & mb & 1) != 0) continue;            // e0*e0 = 0
      const int k  = M2I_[ma ^ mb];
      const int sg = rsign_(ma, mb);
      U[k] += (sg > 0) ? (A[i]*B[j]) : (-(A[i]*B[j]));
    }
  }
}

// ---------------- weight pre-swizzle ----------------
__device__ void pack_equi(const float* w, int Cout, _Float16* d, int ntn){
  for(int g=0; g<5; ++g) for(int nt=0; nt<ntn; ++nt){
    _Float16* td = d + (g*ntn + nt)*512;
    for(int pos = threadIdx.x; pos < 512; pos += blockDim.x){
      int lane = pos>>4, i = pos&15;
      int N = nt*16 + (lane & 15);
      int K = i + ((lane >> 4) << 4);             // 0..31, valid K<16
      float v = (K < 16) ? w[(g*Cout + N)*16 + K] : 0.f;
      td[pos] = (_Float16)v;
    }
  }
}
__device__ void pack_mat(const float* w, int Kin, int Nout, _Float16* d){
  int ntn = Nout >> 4, nkc = Kin >> 5;
  for(int nt=0; nt<ntn; ++nt) for(int kc=0; kc<nkc; ++kc){
    _Float16* td = d + (nt*nkc + kc)*512;
    for(int pos = threadIdx.x; pos < 512; pos += blockDim.x){
      int lane = pos>>4, i = pos&15;
      int N = nt*16 + (lane & 15);
      int K = kc*32 + i + ((lane >> 4) << 4);
      td[pos] = (_Float16)w[K*Nout + N];
    }
  }
}
__global__ __launch_bounds__(256) void pack_kernel(BlockP bp, _Float16* d){
  switch(blockIdx.x){
    case 0:  pack_equi(bp.wq, 16, d+PK_WQ, 1); break;
    case 1:  pack_equi(bp.wk, 16, d+PK_WK, 1); break;
    case 2:  pack_equi(bp.wv, 16, d+PK_WV, 1); break;
    case 3:  pack_equi(bp.wo, 16, d+PK_WO, 1); break;
    case 4:  pack_equi(bp.w1, 32, d+PK_W1, 2); break;
    case 5:  pack_equi(bp.w2, 16, d+PK_W2, 1); break;
    case 6:  pack_mat(bp.sq,  64,  64, d+PK_SQ); break;
    case 7:  pack_mat(bp.sk,  64,  64, d+PK_SK); break;
    case 8:  pack_mat(bp.sv,  64,  64, d+PK_SV); break;
    case 9:  pack_mat(bp.so,  64,  64, d+PK_SO); break;
    case 10: pack_mat(bp.s1,  64, 128, d+PK_S1); break;
    case 11: pack_mat(bp.s2, 128,  64, d+PK_S2); break;
    default: break;
  }
}

// ---------------- embedding ----------------
__global__ __launch_bounds__(256) void embed_kernel(const float* verts, const int* faces,
                                                    const float* w_in, float* x, float* s){
  int gid = blockIdx.x*256 + threadIdx.x;   // TN*16 threads
  int t = gid >> 4, o = gid & 15;
  float mv[16];
#pragma unroll
  for(int k=0;k<16;++k) mv[k]=0.f;
  int c;
  if(t < NV){
    c = 0;
    float p0 = verts[t*3+0], p1 = verts[t*3+1], p2 = verts[t*3+2];
    mv[14] = 1.f; mv[13] = -p0; mv[12] = p1; mv[11] = -p2;
  } else {
    c = 1;
    int f = t - NV;
    int i0 = faces[f*3+0], i1 = faces[f*3+1], i2 = faces[f*3+2];
    float a0=verts[i0*3],a1=verts[i0*3+1],a2=verts[i0*3+2];
    float b0=verts[i1*3],b1=verts[i1*3+1],b2=verts[i1*3+2];
    float d0=verts[i2*3],d1=verts[i2*3+1],d2=verts[i2*3+2];
    float cx=(a0+b0+d0)*(1.f/3.f), cy=(a1+b1+d1)*(1.f/3.f), cz=(a2+b2+d2)*(1.f/3.f);
    float e10=b0-a0,e11=b1-a1,e12=b2-a2;
    float e20=d0-a0,e21=d1-a1,e22=d2-a2;
    float nx=e11*e22-e12*e21, ny=e12*e20-e10*e22, nz=e10*e21-e11*e20;
    float nn = sqrtf(nx*nx+ny*ny+nz*nz) + 1e-8f;
    nx/=nn; ny/=nn; nz/=nn;
    mv[2]=nx; mv[3]=ny; mv[4]=nz;
    mv[1]=cx*nx + cy*ny + cz*nz;
  }
  float* xr = x + (size_t)t*256 + o*16;
#pragma unroll
  for(int k=0;k<16;++k) xr[k] = mv[k] * w_in[(GRADE_[k]*16 + o)*2 + c];
  if(o < 4){
    float* sr = s + (size_t)t*64 + o*16;
#pragma unroll
    for(int i=0;i<16;++i) sr[i]=0.f;
  }
}

// ---------------- shared norm builder ----------------
__device__ void build_xn_sn(const float* x, const float* s, int tb,
                            _Float16 (*xn)[16][32], _Float16 (*sn)[64], float* rinv,
                            const float* lng, const float* lnb)
{
  int lane = (int)(threadIdx.x & 31u);
  // zero channel padding 16..31
  for(int idx=lane; idx<16*16*16; idx+=32){
    int bl=idx>>8, t=(idx>>4)&15, c=idx&15;
    xn[bl][t][16+c] = (_Float16)0.f;
  }
  if(lane < 16){
    const float* xr = x + (size_t)(tb+lane)*256;
    float n = 0.f;
    for(int i=0;i<256;++i){ float v = xr[i]; n += v*v*INVM_[i&15]; }
    rinv[lane] = rsqrtf(n*(1.f/16.f) + 1e-6f);
    const float* sr = s + (size_t)(tb+lane)*64;
    float mu=0.f; for(int i=0;i<64;++i) mu += sr[i]; mu *= (1.f/64.f);
    float var=0.f; for(int i=0;i<64;++i){ float d=sr[i]-mu; var += d*d; } var *= (1.f/64.f);
    float is = rsqrtf(var + 1e-6f);
    for(int i=0;i<64;++i) sn[lane][i] = (_Float16)((sr[i]-mu)*is*lng[i] + lnb[i]);
  }
  // blade-major normalized mv into LDS (A-layout friendly)
  for(int p=lane; p<256; p+=32){
    int t=p>>4, c=p&15;
    const float* xr = x + (size_t)(tb+t)*256 + c*16;
    float ri = rinv[t];
#pragma unroll
    for(int k=0;k<16;++k) xn[k][t][c] = (_Float16)(xr[k]*ri);
  }
}

// ---------------- QKV projection + head packing ----------------
__global__ __launch_bounds__(64) void qkv_kernel(const float* x, const float* s, BlockP bp,
                                                 const _Float16* wp,
                                                 _Float16* Qp, _Float16* Kp, _Float16* Vt)
{
  __shared__ _Float16 xnS[2][16][16][32];
  __shared__ _Float16 snS[2][16][64];
  __shared__ float    rinvS[2][16];
  int wave = threadIdx.x >> 5, lane = threadIdx.x & 31;
  int tile = blockIdx.x*2 + wave, tb = tile*16;
  build_xn_sn(x, s, tb, xnS[wave], snS[wave], rinvS[wave], bp.ln1_g, bp.ln1_b);

  // zero the padded feature region 40..63 (Qp/Kp rows, Vt feature rows)
  {
    v8h z = {};
    for(int idx=lane; idx<8*16; idx+=32){
      int h = idx>>4, t = tb + (idx & 15);
      _Float16* q = Qp + ((size_t)h*TN + t)*64 + 40;
      _Float16* k = Kp + ((size_t)h*TN + t)*64 + 40;
      *(v8h*)q = z; *(v8h*)(q+8) = z; *(v8h*)(q+16) = z;
      *(v8h*)k = z; *(v8h*)(k+8) = z; *(v8h*)(k+16) = z;
    }
    for(int idx=lane; idx<8*24; idx+=32){
      int h = idx/24, f = 40 + idx%24;
      _Float16* v = Vt + ((size_t)h*64 + f)*TN + tb;
      *(v8h*)v = z; *(v8h*)(v+8) = z;
    }
  }
  const float scale = 0.15811388300841898f;  // 1/sqrt(CH*8 + SH) = 1/sqrt(40)
  int g = lane>>4, n = lane&15;
  // multivector q/k/v : per-blade 16x16x16 matmuls on WMMA
  for(int k=0;k<16;++k){
    v16h a = loadA16(&xnS[wave][k][0][0], 32, 0);
    int gr = GRADE_[k];
    v8f zq = {}, zk = {}, zv = {};
    zq = wmma32(a, loadBpk(wp + PK_WQ + gr*512), zq);
    zk = wmma32(a, loadBpk(wp + PK_WK + gr*512), zk);
    zv = wmma32(a, loadBpk(wp + PK_WV + gr*512), zv);
    float im = INVM_[k]*scale;
#pragma unroll
    for(int j=0;j<8;++j){
      int t = tb + j + (g<<3);
      int h = n>>1, f = (n&1)*16 + k;
      Qp[((size_t)h*TN + t)*64 + f] = (_Float16)(zq[j]*im);
      Kp[((size_t)h*TN + t)*64 + f] = (_Float16)zk[j];
      Vt[((size_t)h*64 + f)*TN + t] = (_Float16)zv[j];
    }
  }
  // scalar q/k/v : 64x64 matmuls on WMMA
  v16h sa0 = loadA16(&snS[wave][0][0], 64, 0);
  v16h sa1 = loadA16(&snS[wave][0][0], 64, 32);
  for(int nt=0; nt<4; ++nt){
    v8f zq = {}, zk = {}, zv = {};
    zq = wmma32(sa0, loadBpk(wp + PK_SQ + (nt*2+0)*512), zq);
    zq = wmma32(sa1, loadBpk(wp + PK_SQ + (nt*2+1)*512), zq);
    zk = wmma32(sa0, loadBpk(wp + PK_SK + (nt*2+0)*512), zk);
    zk = wmma32(sa1, loadBpk(wp + PK_SK + (nt*2+1)*512), zk);
    zv = wmma32(sa0, loadBpk(wp + PK_SV + (nt*2+0)*512), zv);
    zv = wmma32(sa1, loadBpk(wp + PK_SV + (nt*2+1)*512), zv);
#pragma unroll
    for(int j=0;j<8;++j){
      int t = tb + j + (g<<3);
      int o = nt*16 + n; int h = o>>3, f = 32 + (o&7);
      Qp[((size_t)h*TN + t)*64 + f] = (_Float16)(zq[j]*scale);
      Kp[((size_t)h*TN + t)*64 + f] = (_Float16)zk[j];
      Vt[((size_t)h*64 + f)*TN + t] = (_Float16)zv[j];
    }
  }
}

// ---------------- flash attention (one wave = one head x 16-query tile) ----------------
__global__ __launch_bounds__(128) void attn_kernel(const _Float16* Qp, const _Float16* Kp,
                                                   const _Float16* Vt, float* OutA)
{
  __shared__ _Float16 plds[4][512];
  int wave = threadIdx.x >> 5, lane = threadIdx.x & 31;
  int gidx = blockIdx.x*4 + wave;          // 0..2047
  int h  = gidx >> 8;
  int qt = gidx & 255;
  const _Float16* Qh = Qp + (size_t)h*TN*64;
  const _Float16* Kh = Kp + (size_t)h*TN*64;
  const _Float16* Vh = Vt + (size_t)h*64*TN;
  int g = lane>>4, n = lane&15;

  v16h aq0 = loadA16(Qh + (size_t)qt*16*64, 64, 0);
  v16h aq1 = loadA16(Qh + (size_t)qt*16*64, 64, 32);

  float m[8], ssum[8];
  v8f acc[3];
  v8f vzero = {};
#pragma unroll
  for(int j=0;j<8;++j){ m[j] = -1e30f; ssum[j] = 0.f; }
#pragma unroll
  for(int i=0;i<3;++i) acc[i] = vzero;

  _Float16* pl = &plds[wave][0];

  // prefetch first K/V tiles (global_prefetch_b8; one 128B line per lane)
  __builtin_prefetch(Kh + (size_t)lane*64, 0, 1);
  __builtin_prefetch(Vh + (size_t)lane*TN, 0, 1);
  __builtin_prefetch(Vh + (size_t)(lane+16)*TN, 0, 1);

  for(int kt=0; kt<TN/32; ++kt){
    int kb = kt*32;
    // prefetch next tile while this one computes
    if(kt+1 < TN/32){
      const _Float16* nK = Kh + (size_t)(kb+32+lane)*64;       // next K tile: 32 rows x 128B
      __builtin_prefetch(nK, 0, 1);
      const _Float16* nV0 = Vh + (size_t)lane*TN      + kb+32; // next V columns, rows 0..31
      const _Float16* nV1 = Vh + (size_t)(lane+16)*TN + kb+32; // rows 16..47
      __builtin_prefetch(nV0, 0, 1);
      __builtin_prefetch(nV1, 0, 1);
    }
    v8f cs[2];
#pragma unroll
    for(int st=0; st<2; ++st){
      const _Float16* kbase = Kh + (size_t)(kb + st*16)*64;
      v8f z = {};
      z = wmma32(aq0, loadB16(kbase, 64, 0),  z);
      z = wmma32(aq1, loadB16(kbase, 64, 32), z);
      cs[st] = z;
    }
    // online softmax (per query row j; rows striped N-wise over 16 lanes)
#pragma unroll
    for(int j=0;j<8;++j){
      float tmax = redmax16(fmaxf(cs[0][j], cs[1][j]));
      float mn   = fmaxf(m[j], tmax);
      float fac  = __expf(m[j] - mn);
      m[j] = mn;
      float p0 = __expf(cs[0][j] - mn);
      float p1 = __expf(cs[1][j] - mn);
      cs[0][j] = p0; cs[1][j] = p1;
      float rs = redsum16(p0 + p1);
      ssum[j] = ssum[j]*fac + rs;
      acc[0][j] *= fac; acc[1][j] *= fac; acc[2][j] *= fac;
    }
    // P: C-layout -> LDS row-major [16q][32k] -> A-layout
#pragma unroll
    for(int j=0;j<8;++j){
      pl[((j + (g<<3))<<5) + n]      = (_Float16)cs[0][j];
      pl[((j + (g<<3))<<5) + 16 + n] = (_Float16)cs[1][j];
    }
    v16h ap = loadA16(pl, 32, 0);
#pragma unroll
    for(int ft=0; ft<3; ++ft){
      v16h bv = loadB16(Vh + (size_t)ft*16*TN, (size_t)TN, kb);
      acc[ft] = wmma32(ap, bv, acc[ft]);
    }
  }
#pragma unroll
  for(int j=0;j<8;++j){
    float inv = 1.f / ssum[j];
    int t = qt*16 + j + (g<<3);
#pragma unroll
    for(int ft=0; ft<3; ++ft)
      OutA[(size_t)t*384 + h*48 + ft*16 + n] = acc[ft][j]*inv;
  }
}

// ---------------- attention output projection ----------------
__global__ __launch_bounds__(64) void attnout_kernel(const float* OutA, BlockP bp,
                                                     const _Float16* wp, float* x, float* s)
{
  __shared__ _Float16 omv[2][16][16][32];
  __shared__ _Float16 osn[2][16][64];
  int wave = threadIdx.x>>5, lane = threadIdx.x&31;
  int tile = blockIdx.x*2 + wave, tb = tile*16;
  int g = lane>>4, n = lane&15;
  for(int idx=lane; idx<16*16*16; idx+=32){
    int bl=idx>>8, t=(idx>>4)&15, c=idx&15;
    omv[wave][bl][t][16+c] = (_Float16)0.f;
  }
  for(int p=lane; p<256; p+=32){
    int t=p>>4, c=p&15;
    const float* r = OutA + (size_t)(tb+t)*384 + (c>>1)*48 + (c&1)*16;
#pragma unroll
    for(int k=0;k<16;++k) omv[wave][k][t][c] = (_Float16)r[k];
  }
  for(int p=lane; p<1024; p+=32){
    int t=p>>6, d=p&63;
    osn[wave][t][d] = (_Float16)OutA[(size_t)(tb+t)*384 + (d>>3)*48 + 32 + (d&7)];
  }
  for(int k=0;k<16;++k){
    v16h a = loadA16(&omv[wave][k][0][0], 32, 0);
    v8f z = {};
    z = wmma32(a, loadBpk(wp + PK_WO + GRADE_[k]*512), z);
#pragma unroll
    for(int j=0;j<8;++j){
      int t = tb + j + (g<<3);
      x[(size_t)t*256 + n*16 + k] += z[j];
    }
  }
  v16h a0 = loadA16(&osn[wave][0][0], 64, 0);
  v16h a1 = loadA16(&osn[wave][0][0], 64, 32);
  for(int nt=0; nt<4; ++nt){
    v8f z = {};
    z = wmma32(a0, loadBpk(wp + PK_SO + (nt*2+0)*512), z);
    z = wmma32(a1, loadBpk(wp + PK_SO + (nt*2+1)*512), z);
#pragma unroll
    for(int j=0;j<8;++j){
      int t = tb + j + (g<<3);
      int o = nt*16 + n;
      s[(size_t)t*64 + o] += z[j] + bp.so_b[o];
    }
  }
}

// ---------------- MLP block (equi w1 -> geometric product -> gated gelu -> equi w2; scalar MLP) ----------------
__global__ __launch_bounds__(32) void mlp_kernel(float* x, float* s, BlockP bp, const _Float16* wp)
{
  __shared__ _Float16 xnS[16][16][32];
  __shared__ _Float16 snS[16][64];
  __shared__ _Float16 hb[16][32][16];
  __shared__ _Float16 hs[16][128];
  __shared__ float    rinvS[16];
  int lane = threadIdx.x & 31;
  int tb = blockIdx.x*16;
  int g = lane>>4, n = lane&15;
  build_xn_sn(x, s, tb, xnS, snS, rinvS, bp.ln2_g, bp.ln2_b);

  // w1: per-blade 16x(32)x16 -> hb[t][c32][blade]
  for(int k=0;k<16;++k){
    v16h a = loadA16(&xnS[k][0][0], 32, 0);
    for(int nt=0; nt<2; ++nt){
      v8f z = {};
      z = wmma32(a, loadBpk(wp + PK_W1 + (GRADE_[k]*2 + nt)*512), z);
#pragma unroll
      for(int j=0;j<8;++j) hb[j + (g<<3)][nt*16 + n][k] = (_Float16)z[j];
    }
  }
  // geometric product + gated gelu -> overwrite xnS (padding still zero)
  for(int p=lane; p<256; p+=32){
    int t=p>>4, c=p&15;
    float A[16], B[16], U[16];
    const _Float16* pa = &hb[t][c][0];
    const _Float16* pb = &hb[t][c+16][0];
#pragma unroll
    for(int i=0;i<16;++i){ A[i]=(float)pa[i]; B[i]=(float)pb[i]; U[i]=0.f; }
    geo_product(A, B, U);
    float gate = gelu_(U[0]);
#pragma unroll
    for(int k=0;k<16;++k) xnS[k][t][c] = (_Float16)(U[k]*gate);
  }
  // w2 -> x residual
  for(int k=0;k<16;++k){
    v16h a = loadA16(&xnS[k][0][0], 32, 0);
    v8f z = {};
    z = wmma32(a, loadBpk(wp + PK_W2 + GRADE_[k]*512), z);
#pragma unroll
    for(int j=0;j<8;++j){
      int t = tb + j + (g<<3);
      x[(size_t)t*256 + n*16 + k] += z[j];
    }
  }
  // scalar MLP: 64 -> 128 (gelu) -> 64
  v16h a0 = loadA16(&snS[0][0], 64, 0);
  v16h a1 = loadA16(&snS[0][0], 64, 32);
  for(int nt=0; nt<8; ++nt){
    v8f z = {};
    z = wmma32(a0, loadBpk(wp + PK_S1 + (nt*2+0)*512), z);
    z = wmma32(a1, loadBpk(wp + PK_S1 + (nt*2+1)*512), z);
#pragma unroll
    for(int j=0;j<8;++j){
      int o = nt*16 + n;
      hs[j + (g<<3)][o] = (_Float16)gelu_(z[j] + bp.s1_b[o]);
    }
  }
  for(int nt=0; nt<4; ++nt){
    v8f z = {};
    for(int kc=0; kc<4; ++kc){
      v16h a = loadA16(&hs[0][0], 128, kc*32);
      z = wmma32(a, loadBpk(wp + PK_S2 + (nt*4+kc)*512), z);
    }
#pragma unroll
    for(int j=0;j<8;++j){
      int t = tb + j + (g<<3);
      int o = nt*16 + n;
      s[(size_t)t*64 + o] += z[j] + bp.s2_b[o];
    }
  }
}

// ---------------- pooling + classifier ----------------
__global__ __launch_bounds__(256) void pool_kernel(const float* x, const float* w_out, float* pool){
  __shared__ float red[256];
  int tid = threadIdx.x;
  int k = tid & 15, seg = tid >> 4;
  int t0 = blockIdx.x*64 + seg*4;
  float acc = 0.f;
  for(int tt=0; tt<4; ++tt){
    const float* xr = x + (size_t)(t0+tt)*256;
    float o = 0.f;
#pragma unroll
    for(int c=0;c<16;++c) o += xr[c*16 + k] * w_out[GRADE_[k]*16 + c];
    acc += o;
  }
  red[tid] = acc;
  __syncthreads();
  for(int st=8; st>=1; st>>=1){
    if(seg < st) red[tid] += red[tid + st*16];
    __syncthreads();
  }
  if(seg == 0) pool[blockIdx.x*16 + k] = red[k];
}

__global__ __launch_bounds__(64) void cls_kernel(const float* pool, const float* cw1, const float* cb1,
                                                 const float* cw2, const float* cb2, float* out){
  __shared__ float pooled[16];
  __shared__ float hc[64];
  int tid = threadIdx.x;
  if(tid < 16){
    float a = 0.f;
    for(int b=0;b<64;++b) a += pool[b*16 + tid];
    pooled[tid] = a * (1.f/4096.f);
  }
  __syncthreads();
  {
    float a = cb1[tid];
#pragma unroll
    for(int k=0;k<16;++k) a += pooled[k]*cw1[k*64 + tid];
    hc[tid] = gelu_(a);
  }
  __syncthreads();
  if(tid < 7){
    float a = cb2[tid];
    for(int d=0;d<64;++d) a += hc[d]*cw2[d*7 + tid];
    out[tid] = a;
  }
}

// ---------------- host launcher ----------------
extern "C" void kernel_launch(void* const* d_in, const int* in_sizes, int n_in,
                              void* d_out, int out_size, void* d_ws, size_t ws_size,
                              hipStream_t stream)
{
  (void)out_size;
  const float* verts; const int* faces;
  BlockP bp[8];
  const float *w_in,*w_out,*cls_w1,*cls_b1,*cls_w2,*cls_b2;
  auto F = [&](int i){ return (const float*)d_in[i]; };

  bool alpha = (n_in >= 160) && (in_sizes[1] == 64);   // jax tree_flatten (alphabetical) vs insertion order
  if(alpha){
    faces = (const int*)d_in[0];
    for(int b=0;b<8;++b){
      int a = 1 + b*19;
      bp[b].ln1_b=F(a+0); bp[b].ln1_g=F(a+1); bp[b].ln2_b=F(a+2); bp[b].ln2_g=F(a+3);
      bp[b].s1=F(a+4); bp[b].s1_b=F(a+5); bp[b].s2=F(a+6); bp[b].s2_b=F(a+7);
      bp[b].sk=F(a+8); bp[b].so=F(a+9); bp[b].so_b=F(a+10); bp[b].sq=F(a+11); bp[b].sv=F(a+12);
      bp[b].w1=F(a+13); bp[b].w2=F(a+14); bp[b].wk=F(a+15); bp[b].wo=F(a+16); bp[b].wq=F(a+17); bp[b].wv=F(a+18);
    }
    cls_b1=F(153); cls_b2=F(154); cls_w1=F(155); cls_w2=F(156); w_in=F(157); w_out=F(158);
    verts=F(159);
  } else {
    verts=F(0); faces=(const int*)d_in[1];
    for(int b=0;b<8;++b){
      int a = 2 + b*19;
      bp[b].wq=F(a+0); bp[b].wk=F(a+1); bp[b].wv=F(a+2); bp[b].wo=F(a+3);
      bp[b].sq=F(a+4); bp[b].sk=F(a+5); bp[b].sv=F(a+6); bp[b].so=F(a+7); bp[b].so_b=F(a+8);
      bp[b].w1=F(a+9); bp[b].w2=F(a+10); bp[b].s1=F(a+11); bp[b].s1_b=F(a+12);
      bp[b].s2=F(a+13); bp[b].s2_b=F(a+14);
      bp[b].ln1_g=F(a+15); bp[b].ln1_b=F(a+16); bp[b].ln2_g=F(a+17); bp[b].ln2_b=F(a+18);
    }
    w_in=F(154); w_out=F(155); cls_w1=F(156); cls_b1=F(157); cls_w2=F(158); cls_b2=F(159);
  }

  // scratch carve
  size_t off = 0;
  auto carve = [&](size_t bytes)->char*{
    char* p = (char*)d_ws + off;
    off = (off + bytes + 255) & ~(size_t)255;
    return p;
  };
  float*     x     = (float*)    carve((size_t)TN*256*4);
  float*     s     = (float*)    carve((size_t)TN*64*4);
  _Float16*  Qp    = (_Float16*) carve((size_t)8*TN*64*2);
  _Float16*  Kp    = (_Float16*) carve((size_t)8*TN*64*2);
  _Float16*  Vt    = (_Float16*) carve((size_t)8*64*TN*2);
  float*     OutA  = (float*)    carve((size_t)TN*384*4);
  _Float16*  wpack = (_Float16*) carve((size_t)8*PK_PER_BLOCK*2);
  float*     pool  = (float*)    carve((size_t)64*16*4);
  if(off > ws_size) return;

  for(int b=0;b<8;++b)
    pack_kernel<<<12, 256, 0, stream>>>(bp[b], wpack + (size_t)b*PK_PER_BLOCK);

  embed_kernel<<<256, 256, 0, stream>>>(verts, faces, w_in, x, s);

  for(int b=0;b<8;++b){
    const _Float16* wb = wpack + (size_t)b*PK_PER_BLOCK;
    qkv_kernel    <<<128, 64,  0, stream>>>(x, s, bp[b], wb, Qp, Kp, Vt);
    attn_kernel   <<<512, 128, 0, stream>>>(Qp, Kp, Vt, OutA);
    attnout_kernel<<<128, 64,  0, stream>>>(OutA, bp[b], wb, x, s);
    mlp_kernel    <<<256, 32,  0, stream>>>(x, s, bp[b], wb);
  }

  pool_kernel<<<64, 256, 0, stream>>>(x, w_out, pool);
  cls_kernel <<<1, 64, 0, stream>>>(pool, cls_w1, cls_b1, cls_w2, cls_b2, (float*)d_out);
}
